// MultiHeadAttention_60619168416296
// MI455X (gfx1250) — matrix operand order repaired and checked by reading, not compiled, not run on hardware
//
#include <hip/hip_runtime.h>
#include <hip/hip_bf16.h>

// ---- problem constants (from reference) ----
#define BATCH   2
#define L_SEQ   2048
#define DMODEL  512
#define NHEAD   8
#define DHEAD   64
#define PADROWS (L_SEQ + 64)   // padded rel tables (zero tail)

typedef __bf16 bf16;
typedef __attribute__((ext_vector_type(16))) __bf16 bf16x16;
typedef __attribute__((ext_vector_type(8)))  __bf16 bf16x8;
typedef __attribute__((ext_vector_type(8)))  float  f32x8;
typedef __attribute__((ext_vector_type(4)))  unsigned int u32x4;
typedef __attribute__((ext_vector_type(8)))  int i32x8;
typedef __attribute__((ext_vector_type(4)))  int i32x4;

// ---- TDM availability (probe-guarded; cooperative-copy fallback) ----
#if defined(__AMDGCN__) && __has_builtin(__builtin_amdgcn_tensor_load_to_lds) && \
    __has_builtin(__builtin_amdgcn_s_wait_tensorcnt)
#define HAVE_TDM 1
#else
#define HAVE_TDM 0
#endif

// ---- WMMA wrapper: D = A(16x32 bf16) x B(32x16 bf16) + C(16x16 f32) ----
__device__ __forceinline__ f32x8 wmma_bf16(bf16x16 a, bf16x16 b, f32x8 c) {
  return __builtin_amdgcn_wmma_f32_16x16x32_bf16(
      false, a, false, b, (short)0, c, false, false);
}

// ---- fragment index maps (wave32; m/n as in the ISA, k in contiguous order, not the ISA table order) ----
__device__ __forceinline__ int a_frag_m(int lane) { return lane & 15; }
__device__ __forceinline__ int a_frag_k(int lane, int e) {
  return ((lane >> 4) << 4) + e;
}
__device__ __forceinline__ int b_frag_n(int lane) { return lane & 15; }
// B: per lane (n fixed), k = (lane>>4)*16 + e  -> 16 CONTIGUOUS k values
// A: per lane (m fixed), k = (lane>>4)*16 + e  -> 16 CONTIGUOUS k values

__device__ __forceinline__ bf16x16 combine8(bf16x8 lo, bf16x8 hi) {
  bf16x16 r;
#pragma unroll
  for (int e = 0; e < 8; ++e) { r[e] = lo[e]; r[e + 8] = hi[e]; }
  return r;
}
// B fragment: 16 contiguous bf16 at p (K-innermost source)
__device__ __forceinline__ bf16x16 load_bfrag(const bf16* p) {
  return combine8(*(const bf16x8*)p, *(const bf16x8*)(p + 8));
}
// A fragment: row pointer at k-base; runs at +g*16 and +g*16+8
__device__ __forceinline__ bf16x16 load_afrag(const bf16* row, int g) {
  return combine8(*(const bf16x8*)(row + g * 16), *(const bf16x8*)(row + g * 16 + 8));
}
__device__ __forceinline__ bf16x16 load_afrag_f32(const float* row, int g) {
  f32x8 lo = *(const f32x8*)(row + g * 16);
  f32x8 hi = *(const f32x8*)(row + g * 16 + 8);
  bf16x16 r;
#pragma unroll
  for (int e = 0; e < 8; ++e) { r[e] = (bf16)lo[e]; r[e + 8] = (bf16)hi[e]; }
  return r;
}

// ---- Tensor Data Mover: 2D tile (rows x width elems) -> LDS with pitch ----
// Optional LDS padding: rows land every lds_pitch_bytes (pad region untouched).
__device__ __forceinline__ void tdm_load_tile(const void* gsrc, void* ldst,
                                              int dsz_code, int width, int rows,
                                              int gstride, int lds_pitch_bytes) {
#if HAVE_TDM
  const int rowbytes = width << dsz_code;
  const int pad_en = (lds_pitch_bytes > rowbytes) ? 1 : 0;
  const int pad_int_code = pad_en ? (__builtin_ctz(rowbytes >> 2) - 1) : 0;  // 2<<c DW
  const int pad_amt_code = pad_en ? (((lds_pitch_bytes - rowbytes) >> 2) - 1) : 0;
  const unsigned long long ga = (unsigned long long)gsrc;
  u32x4 g0;
  g0[0] = 1u;                                                    // count=1, user D#
  g0[1] = (unsigned int)(unsigned long long)ldst;                // lds_addr (bytes)
  g0[2] = (unsigned int)ga;                                      // global_addr lo
  g0[3] = ((unsigned int)(ga >> 32) & 0x01FFFFFFu) | (2u << 30); // addr hi | type=2
  i32x8 g1;
  g1[0] = (dsz_code << 16) | (pad_en << 20) | (pad_int_code << 22) | (pad_amt_code << 25);
  g1[1] = (width & 0xFFFF) << 16;                                // tensor_dim0 lo16
  g1[2] = ((width >> 16) & 0xFFFF) | ((rows & 0xFFFF) << 16);    // dim0 hi | dim1 lo
  g1[3] = ((rows >> 16) & 0xFFFF) | ((width & 0xFFFF) << 16);    // dim1 hi | tile_dim0
  g1[4] = rows & 0xFFFF;                                         // tile_dim1
  g1[5] = gstride;                                               // dim0_stride lo32
  g1[6] = 0;
  g1[7] = 0;
  i32x4 z4 = {0, 0, 0, 0};
#if __clang_major__ >= 23
  i32x8 z8 = {0, 0, 0, 0, 0, 0, 0, 0};
  __builtin_amdgcn_tensor_load_to_lds(g0, g1, z4, z4, z8, 0);
#else
  __builtin_amdgcn_tensor_load_to_lds(g0, g1, z4, z4, 0);
#endif
#else
  const int lane = (int)(threadIdx.x & 31);
  const int esz = 1 << dsz_code;
  for (int r = lane; r < rows; r += 32) {
    const char* s = (const char*)gsrc + (long long)r * gstride * esz;
    char* d = (char*)ldst + (long long)r * lds_pitch_bytes;
    for (int c = 0; c < width * esz; c += 4)
      *(int*)(d + c) = *(const int*)(s + c);
  }
#endif
}
__device__ __forceinline__ void tdm_wait() {
#if HAVE_TDM
  __builtin_amdgcn_s_wait_tensorcnt(0);
#endif
}

// =====================================================================
// Weight transpose+convert: wT[n][k] = (bf16) w[k][n]
// =====================================================================
__global__ void transpose_w(const float* __restrict__ w, bf16* __restrict__ wT) {
  const int idx = blockIdx.x * blockDim.x + threadIdx.x;
  if (idx >= DMODEL * DMODEL) return;
  const int n = idx / DMODEL, k = idx % DMODEL;
  wT[idx] = (bf16)w[(size_t)k * DMODEL + n];
}

// =====================================================================
// Projection: out = x @ W + bias, head-split.  transposed=0 -> [b,h,l,d]
// (for q,k), transposed=1 -> [b,h,d,l] (for v, feeds AV B-fragments).
// grid = (NHEAD, B*L/16), block = 32 (1 wave).
// =====================================================================
__global__ void proj_head_split(const float* __restrict__ x, const bf16* __restrict__ wT,
                                const float* __restrict__ bias, bf16* __restrict__ ouths,
                                int transposed) {
  __shared__ __align__(32) float xlds[16][32];
  __shared__ __align__(32) bf16  wlds[64][32];
  const int lane = threadIdx.x;
  const int h = blockIdx.x, row0 = blockIdx.y * 16;
  const int n0h = h * DHEAD;
  const int am = a_frag_m(lane), bn = b_frag_n(lane), g = lane >> 4;

  f32x8 acc[4] = {};
  for (int k0 = 0; k0 < DMODEL; k0 += 32) {
    tdm_load_tile(x + (size_t)row0 * DMODEL + k0, &xlds[0][0], 2, 32, 16, DMODEL, 32 * 4);
    tdm_load_tile(wT + (size_t)n0h * DMODEL + k0, &wlds[0][0], 1, 32, 64, DMODEL, 32 * 2);
    tdm_wait();
    const bf16x16 afrag = load_afrag_f32(&xlds[am][0], g);
#pragma unroll
    for (int t = 0; t < 4; ++t)
      acc[t] = wmma_bf16(afrag, load_bfrag(&wlds[t * 16 + bn][g * 16]), acc[t]);
    __syncthreads();
  }

#pragma unroll
  for (int t = 0; t < 4; ++t) {
    const int n = t * 16 + bn;
    const float bv = bias[n0h + n];
#pragma unroll
    for (int v = 0; v < 8; ++v) {
      const int row = row0 + v + 8 * g;
      const int b = row / L_SEQ, l = row % L_SEQ;
      const bf16 val = (bf16)(acc[t][v] + bv);
      if (transposed)
        ouths[(((size_t)b * NHEAD + h) * DHEAD + n) * L_SEQ + l] = val;
      else
        ouths[(((size_t)b * NHEAD + h) * L_SEQ + l) * DHEAD + n] = val;
    }
  }
}

// =====================================================================
// Rel tables: krp[row][d] (K-innermost for rel GEMM), vrpT[d][row]
// (row-innermost for skew-value GEMM). Zero tail padding.
// =====================================================================
__global__ void pad_rel(const float* __restrict__ kr, const float* __restrict__ vr,
                        bf16* __restrict__ krp, bf16* __restrict__ vrpT) {
  const int idx = blockIdx.x * blockDim.x + threadIdx.x;
  if (idx >= PADROWS * DHEAD) return;
  const int row = idx / DHEAD, d = idx % DHEAD;
  float a = 0.f, b = 0.f;
  if (row < L_SEQ) { a = kr[idx]; b = vr[idx]; }
  krp[idx] = (bf16)a;
  vrpT[(size_t)d * PADROWS + row] = (bf16)b;
}

// =====================================================================
// C1: logits tile = QK^T + de-skewed rel term, causal mask. TDM-staged.
// grid = (L/32, L/16, B*H), block = 32.
// =====================================================================
__global__ void logits_kernel(const bf16* __restrict__ qh, const bf16* __restrict__ kh,
                              const bf16* __restrict__ krp, float* __restrict__ aw) {
  __shared__ __align__(32) bf16 qlds[16][64];
  __shared__ __align__(32) bf16 khlds[32][64];
  __shared__ __align__(32) bf16 krlds[48][64];
  __shared__ __align__(32) float Tld[16 * 48];
  const int lane = threadIdx.x;
  const int jt = blockIdx.x, it = blockIdx.y, bh = blockIdx.z;
  const int i0 = it * 16, j0 = jt * 32;
  if (j0 > i0 + 15) return;

  const int base0 = (L_SEQ - 16) - i0 + j0;        // >=0, base0+47 < PADROWS
  tdm_load_tile(qh + ((size_t)bh * L_SEQ + i0) * DHEAD, &qlds[0][0], 1, 64, 16, 64, 128);
  tdm_load_tile(kh + ((size_t)bh * L_SEQ + j0) * DHEAD, &khlds[0][0], 1, 64, 32, 64, 128);
  tdm_load_tile(krp + (size_t)base0 * DHEAD, &krlds[0][0], 1, 64, 48, 64, 128);
  tdm_wait();

  const int am = a_frag_m(lane), bn = b_frag_n(lane), g = lane >> 4;
  bf16x16 aq[2];
#pragma unroll
  for (int hk = 0; hk < 2; ++hk) aq[hk] = load_afrag(&qlds[am][hk * 32], g);

  // QK^T (K = d): B rows are kh rows, d contiguous per lane
  f32x8 qk[2];
#pragma unroll
  for (int s = 0; s < 2; ++s) {
    f32x8 c = {};
#pragma unroll
    for (int hk = 0; hk < 2; ++hk)
      c = wmma_bf16(aq[hk], load_bfrag(&khlds[s * 16 + bn][hk * 32 + g * 16]), c);
    qk[s] = c;
  }

  // rel T[r,cc], cc in [0,48)
#pragma unroll
  for (int s = 0; s < 3; ++s) {
    f32x8 c = {};
#pragma unroll
    for (int hk = 0; hk < 2; ++hk)
      c = wmma_bf16(aq[hk], load_bfrag(&krlds[s * 16 + bn][hk * 32 + g * 16]), c);
#pragma unroll
    for (int v = 0; v < 8; ++v)
      Tld[(v + 8 * g) * 48 + s * 16 + bn] = c[v];
  }
  __syncthreads();

#pragma unroll
  for (int s = 0; s < 2; ++s)
#pragma unroll
    for (int v = 0; v < 8; ++v) {
      const int r = v + 8 * g;
      const int c = s * 16 + bn;
      float val = qk[s][v] + Tld[r * 48 + (c - r + 15)];
      const int gi = i0 + r, gj = j0 + c;
      if (gj > gi) val = -1e30f;
      aw[((size_t)bh * L_SEQ + gi) * L_SEQ + gj] = val;
    }
}

// =====================================================================
// C2: row softmax over [0,i]; zero (i, L). One 256-thread block per row.
// =====================================================================
__global__ void softmax_kernel(float* __restrict__ aw) {
  const int row = blockIdx.x;
  const int i = row % L_SEQ;
  float* p = aw + (size_t)row * L_SEQ;
  const int tid = threadIdx.x, nt = blockDim.x;
  __shared__ float red[256];

  for (int j = tid * 64; j <= i; j += nt * 64) __builtin_prefetch(p + j, 0, 1);

  float m = -1e30f;
  for (int j = tid; j <= i; j += nt) m = fmaxf(m, p[j]);
  red[tid] = m; __syncthreads();
  for (int s = 128; s > 0; s >>= 1) {
    if (tid < s) red[tid] = fmaxf(red[tid], red[tid + s]);
    __syncthreads();
  }
  m = red[0]; __syncthreads();

  float sum = 0.f;
  for (int j = tid; j <= i; j += nt) {
    const float e = __expf(p[j] - m);
    p[j] = e; sum += e;
  }
  red[tid] = sum; __syncthreads();
  for (int s = 128; s > 0; s >>= 1) {
    if (tid < s) red[tid] += red[tid + s];
    __syncthreads();
  }
  const float inv = 1.0f / red[0];

  for (int j = tid; j <= i; j += nt) p[j] *= inv;
  for (int j = i + 1 + tid; j < L_SEQ; j += nt) p[j] = 0.f;
}

// =====================================================================
// C3: attn = aw @ vh + skew(aw) @ val_rel  -> bf16 [B,H,L,64]
// aw tile TDM-loaded with LDS padding into 96-wide rows whose side bands
// stay zero -> skew gather awlds[m][cc+m+1] is branch-free.
// grid = (L/16, B*H), block = 32.
// =====================================================================
__global__ void attnv_kernel(const float* __restrict__ aw, const bf16* __restrict__ vhT,
                             const bf16* __restrict__ vrpT, bf16* __restrict__ attn) {
  __shared__ __align__(32) float awlds[16][96];   // data cols [16,48)
  const int lane = threadIdx.x;
  const int it = blockIdx.x, bh = blockIdx.y;
  const int i0 = it * 16;
  const int am = a_frag_m(lane), bn = b_frag_n(lane), g = lane >> 4;
  const size_t awbase = ((size_t)bh * L_SEQ + i0) * L_SEQ;
  const size_t vtbase = (size_t)bh * DHEAD * L_SEQ;

  // zero the bands once (TDM never touches them)
  for (int t = lane; t < 16 * 96; t += 32) {
    const int col = t % 96;
    if (col < 16 || col >= 48) (&awlds[0][0])[t] = 0.f;
  }

  f32x8 acc[4] = {};
  for (int j0 = 0; j0 <= i0 + 15; j0 += 32) {
    tdm_load_tile(aw + awbase + j0, &awlds[0][16], 2, 32, 16, L_SEQ, 96 * 4);
    tdm_wait();

    const bf16x16 af = load_afrag_f32(&awlds[am][16], g);
    bf16x16 sf[2];
#pragma unroll
    for (int hk = 0; hk < 2; ++hk)
#pragma unroll
      for (int e = 0; e < 16; ++e) {
        const int cc = hk * 32 + a_frag_k(lane, e);
        sf[hk][e] = (bf16)awlds[am][cc + am + 1];   // zero outside band
      }

    const int base0 = (L_SEQ - 16) - i0 + j0;
#pragma unroll
    for (int t = 0; t < 4; ++t) {
      // aw @ vh : K = j, vhT is [d][l] so k contiguous per lane
      acc[t] = wmma_bf16(af,
          load_bfrag(vhT + vtbase + (size_t)(t * 16 + bn) * L_SEQ + j0 + g * 16), acc[t]);
      // skew(aw) @ val_rel : K = cc, vrpT is [d][row]
#pragma unroll
      for (int hk = 0; hk < 2; ++hk)
        acc[t] = wmma_bf16(sf[hk],
            load_bfrag(vrpT + (size_t)(t * 16 + bn) * PADROWS + base0 + hk * 32 + g * 16),
            acc[t]);
    }
    __syncthreads();
  }

#pragma unroll
  for (int t = 0; t < 4; ++t)
#pragma unroll
    for (int v = 0; v < 8; ++v)
      attn[((size_t)bh * L_SEQ + i0 + v + 8 * g) * DHEAD + t * 16 + bn] = (bf16)acc[t][v];
}

// =====================================================================
// D: merge heads + out = merged @ wo + bo (fp32).  grid=(8, B*L/16).
// =====================================================================
__global__ void out_proj(const bf16* __restrict__ attn, const bf16* __restrict__ woT,
                         const float* __restrict__ bo, float* __restrict__ out) {
  __shared__ __align__(32) bf16 alds[16][32];
  __shared__ __align__(32) bf16 wlds[64][32];
  const int lane = threadIdx.x;
  const int n0 = blockIdx.x * 64, row0 = blockIdx.y * 16;
  const int am = a_frag_m(lane), bn = b_frag_n(lane), g = lane >> 4;
  const int b = row0 / L_SEQ, l0 = row0 % L_SEQ;

  f32x8 acc[4] = {};
  for (int k0 = 0; k0 < DMODEL; k0 += 32) {
    const int h = k0 >> 6, di0 = k0 & 63;          // merge-heads gather (k0 stays in-head)
    tdm_load_tile(attn + (((size_t)b * NHEAD + h) * L_SEQ + l0) * DHEAD + di0,
                  &alds[0][0], 1, 32, 16, DHEAD, 32 * 2);
    tdm_load_tile(woT + (size_t)n0 * DMODEL + k0, &wlds[0][0], 1, 32, 64, DMODEL, 32 * 2);
    tdm_wait();
    const bf16x16 afrag = load_afrag(&alds[am][0], g);
#pragma unroll
    for (int t = 0; t < 4; ++t)
      acc[t] = wmma_bf16(afrag, load_bfrag(&wlds[t * 16 + bn][g * 16]), acc[t]);
    __syncthreads();
  }
#pragma unroll
  for (int t = 0; t < 4; ++t) {
    const int n = n0 + t * 16 + bn;
    const float bv = bo[n];
#pragma unroll
    for (int v = 0; v < 8; ++v)
      out[(size_t)(row0 + v + 8 * g) * DMODEL + n] = acc[t][v] + bv;
  }
}

// =====================================================================
extern "C" void kernel_launch(void* const* d_in, const int* in_sizes, int n_in,
                              void* d_out, int out_size, void* d_ws, size_t ws_size,
                              hipStream_t stream) {
  (void)in_sizes; (void)n_in; (void)out_size; (void)ws_size;
  const float* q       = (const float*)d_in[0];
  const float* k       = (const float*)d_in[1];
  const float* v       = (const float*)d_in[2];
  /* d_in[3] = causal mask, hardcoded */
  const float* wq      = (const float*)d_in[4];
  const float* bq      = (const float*)d_in[5];
  const float* wk      = (const float*)d_in[6];
  const float* bk      = (const float*)d_in[7];
  const float* wv      = (const float*)d_in[8];
  const float* bv      = (const float*)d_in[9];
  const float* wo      = (const float*)d_in[10];
  const float* bo      = (const float*)d_in[11];
  const float* key_rel = (const float*)d_in[12];
  const float* val_rel = (const float*)d_in[13];

  float* out = (float*)d_out;
  float* aw  = out + (size_t)BATCH * L_SEQ * DMODEL;

  char* ws = (char*)d_ws;
  const size_t szh = (size_t)BATCH * NHEAD * L_SEQ * DHEAD * sizeof(bf16);
  const size_t szr = (size_t)PADROWS * DHEAD * sizeof(bf16);
  const size_t szw = (size_t)DMODEL * DMODEL * sizeof(bf16);
  bf16* qh   = (bf16*)(ws);
  bf16* kh   = (bf16*)(ws + szh);
  bf16* vhT  = (bf16*)(ws + 2 * szh);
  bf16* attn = (bf16*)(ws + 3 * szh);
  bf16* krp  = (bf16*)(ws + 4 * szh);
  bf16* vrpT = (bf16*)(ws + 4 * szh + szr);
  bf16* wqT  = (bf16*)(ws + 4 * szh + 2 * szr);
  bf16* wkT  = (bf16*)(ws + 4 * szh + 2 * szr + szw);
  bf16* wvT  = (bf16*)(ws + 4 * szh + 2 * szr + 2 * szw);
  bf16* woT  = (bf16*)(ws + 4 * szh + 2 * szr + 3 * szw);

  const int ntw = (DMODEL * DMODEL + 255) / 256;
  transpose_w<<<ntw, 256, 0, stream>>>(wq, wqT);
  transpose_w<<<ntw, 256, 0, stream>>>(wk, wkT);
  transpose_w<<<ntw, 256, 0, stream>>>(wv, wvT);
  transpose_w<<<ntw, 256, 0, stream>>>(wo, woT);

  const dim3 gproj(NHEAD, BATCH * L_SEQ / 16);
  proj_head_split<<<gproj, 32, 0, stream>>>(q, wqT, bq, qh, 0);
  proj_head_split<<<gproj, 32, 0, stream>>>(k, wkT, bk, kh, 0);
  proj_head_split<<<gproj, 32, 0, stream>>>(v, wvT, bv, vhT, 1);

  pad_rel<<<(PADROWS * DHEAD + 255) / 256, 256, 0, stream>>>(key_rel, val_rel, krp, vrpT);

  logits_kernel<<<dim3(L_SEQ / 32, L_SEQ / 16, BATCH * NHEAD), 32, 0, stream>>>(qh, kh, krp, aw);
  softmax_kernel<<<BATCH * NHEAD * L_SEQ, 256, 0, stream>>>(aw);
  attnv_kernel<<<dim3(L_SEQ / 16, BATCH * NHEAD), 32, 0, stream>>>(aw, vhT, vrpT, attn);
  out_proj<<<dim3(DMODEL / 64, BATCH * L_SEQ / 16), 32, 0, stream>>>(attn, woT, bo, out);
}
